// DifferentialAttention_8538394985202
// MI455X (gfx1250) — matrix operand order repaired and checked by reading, not compiled, not run on hardware
//
#include <hip/hip_runtime.h>
#include <hip/hip_bf16.h>
#include <stdint.h>

#define B_ 2
#define T_ 1024
#define DIM_ 4096
#define H_ 32
#define HK_ 8
#define D_ 128
#define NREP_ 4
#define SCALE_ 0.08838834764831845f
#define LAMBDA_INIT_ 0.2f
#define EPS_ 1e-6f

typedef __attribute__((ext_vector_type(16))) __bf16 v16bf;
typedef __attribute__((ext_vector_type(8)))  __bf16 v8bf;
typedef __attribute__((ext_vector_type(4)))  __bf16 v4bf;
typedef __attribute__((ext_vector_type(8)))  float  v8f;
typedef __attribute__((ext_vector_type(4)))  float  v4f;
typedef __attribute__((ext_vector_type(4)))  int    v4i;

#if __has_builtin(__builtin_amdgcn_global_load_async_to_lds_b128)
#define USE_ASYNC_LDS 1
#else
#define USE_ASYNC_LDS 0
#endif

#if __has_builtin(__builtin_amdgcn_permlane16)
#define HAVE_PERMLANE16 1
#define PERMLANE16(x, s0, s1) __builtin_amdgcn_permlane16((x), (x), (s0), (s1), false, false)
#elif __has_builtin(__builtin_amdgcn_permlane16_b32)
#define HAVE_PERMLANE16 1
#define PERMLANE16(x, s0, s1) __builtin_amdgcn_permlane16_b32((x), (x), (s0), (s1), false, false)
#else
#define HAVE_PERMLANE16 0
#endif

__device__ __forceinline__ __bf16 f2bf(float f) {
  unsigned u = __builtin_bit_cast(unsigned, f);
  unsigned r = u + 0x7FFFu + ((u >> 16) & 1u);  // round-to-nearest-even
  return __builtin_bit_cast(__bf16, (unsigned short)(r >> 16));
}
__device__ __forceinline__ float bf2f(__bf16 h) {
  unsigned u = ((unsigned)__builtin_bit_cast(unsigned short, h)) << 16;
  return __builtin_bit_cast(float, u);
}
__device__ __forceinline__ v8f wmma_bf16(v16bf a, v16bf b, v8f c) {
  return __builtin_amdgcn_wmma_f32_16x16x32_bf16(false, a, false, b, (short)0, c,
                                                 false, false);
}
#define SHUF16(lo, hi) \
  __builtin_shufflevector(lo, hi, 0, 1, 2, 3, 4, 5, 6, 7, 8, 9, 10, 11, 12, 13, 14, 15)

// XOR-butterfly within a 16-lane row via v_permlane16_b32 (pure VALU; no LDS).
// sel nibble j = j ^ M (lanes 0-7 in s0, lanes 8-15 in s1).
template <int M>
__device__ __forceinline__ float shfl_xor16_t(float x) {
#if HAVE_PERMLANE16
  constexpr int s0 = (M == 1)   ? (int)0x67452301
                     : (M == 2) ? (int)0x54761032
                     : (M == 4) ? (int)0x32107654
                                : (int)0xFEDCBA98;
  constexpr int s1 = (M == 1)   ? (int)0xEFCDAB89
                     : (M == 2) ? (int)0xDCFE98BA
                     : (M == 4) ? (int)0xBA98FEDC
                                : (int)0x76543210;
  int xi = __builtin_bit_cast(int, x);
  int r = PERMLANE16(xi, s0, s1);
  return __builtin_bit_cast(float, r);
#else
  return __shfl_xor(x, M, 16);
#endif
}
__device__ __forceinline__ float redmax16(float v) {
  v = fmaxf(v, shfl_xor16_t<1>(v));
  v = fmaxf(v, shfl_xor16_t<2>(v));
  v = fmaxf(v, shfl_xor16_t<4>(v));
  v = fmaxf(v, shfl_xor16_t<8>(v));
  return v;
}
__device__ __forceinline__ float redsum16(float v) {
  v += shfl_xor16_t<1>(v);
  v += shfl_xor16_t<2>(v);
  v += shfl_xor16_t<4>(v);
  v += shfl_xor16_t<8>(v);
  return v;
}

// ---------------------------------------------------------------------------
// Elementwise f32 -> bf16 (4 elems/thread, vector load/store).
// ---------------------------------------------------------------------------
__global__ void convert_bf16(const float* __restrict__ in, __bf16* __restrict__ outp,
                             int total4) {
  int idx = blockIdx.x * blockDim.x + threadIdx.x;
  if (idx >= total4) return;
  v4f v = reinterpret_cast<const v4f*>(in)[idx];
  v4bf o;
#pragma unroll
  for (int i = 0; i < 4; ++i) o[i] = f2bf(v[i]);
  reinterpret_cast<v4bf*>(outp)[idx] = o;
}

// ---------------------------------------------------------------------------
// Tiled transpose + convert: W (K x N, f32) -> Wt (N x K, bf16).
// ---------------------------------------------------------------------------
__global__ __launch_bounds__(256) void transpose_convert(
    const float* __restrict__ W, __bf16* __restrict__ Wt, int K, int N) {
  __shared__ float tile[32][33];
  const int n0 = blockIdx.x * 32, k0 = blockIdx.y * 32;
  const int tx = threadIdx.x, ty = threadIdx.y;
#pragma unroll
  for (int i = ty; i < 32; i += 8)
    tile[i][tx] = W[(size_t)(k0 + i) * N + n0 + tx];
  __syncthreads();
#pragma unroll
  for (int i = ty; i < 32; i += 8)
    Wt[(size_t)(n0 + i) * K + k0 + tx] = f2bf(tile[tx][i]);
}

// ---------------------------------------------------------------------------
// Tiled transpose (bf16): v (B*T, HK, D) -> vT (B, HK, D, T).
// ---------------------------------------------------------------------------
__global__ __launch_bounds__(256) void transpose_v(const __bf16* __restrict__ v,
                                                   __bf16* __restrict__ vT) {
  __shared__ __bf16 tile[32][33];
  const int bh = blockIdx.z, b = bh / HK_, hk = bh % HK_;
  const int t0 = blockIdx.y * 32, d0 = blockIdx.x * 32;
  const int tx = threadIdx.x, ty = threadIdx.y;
#pragma unroll
  for (int i = ty; i < 32; i += 8)
    tile[i][tx] = v[((size_t)(b * T_ + t0 + i) * HK_ + hk) * D_ + d0 + tx];
  __syncthreads();
#pragma unroll
  for (int i = ty; i < 32; i += 8)
    vT[((size_t)(b * HK_ + hk) * D_ + d0 + i) * T_ + t0 + tx] = tile[tx][i];
}

// ---------------------------------------------------------------------------
// GEMM: C[M,N] = A[M,K] @ Bt[N,K]^T ; bf16 operands, f32 WMMA accumulate.
// 256 threads = 8 waves (2x4); wave tile 32(M) x 64(N); block tile 64 x 256.
// Software-pipelined 2-deep: B-fragments ping-pong in VGPRs; the shared
// 64x32 A-tile is double-buffered in LDS, staged with async load-to-LDS
// (ASYNCcnt path) when the toolchain exposes the builtin.
// ---------------------------------------------------------------------------
template <bool OUT_BF16>
__global__ __launch_bounds__(256) void gemm_bf16_wmma(
    const __bf16* __restrict__ A, const __bf16* __restrict__ Bt,
    void* __restrict__ Cout, int M, int N, int K) {
  __shared__ __bf16 atile[2][64][32];  // 8 KB double-buffered A block-tile

  const int tid   = threadIdx.x;
  const int lane  = tid & 31;
  const int ln16  = lane & 15;
  const int khalf = lane >> 4;
  const int wave  = tid >> 5;
  const int wr = wave >> 2, wc = wave & 3;
  const int m0 = blockIdx.y * 64 + wr * 32;
  const int n0 = blockIdx.x * 256 + wc * 64;

  // staging: each thread copies one 16B chunk of the 64x32 A tile
  const int srow = tid >> 2;
  const int sseg = (tid & 3) * 8;
  const __bf16* aStage = A + (size_t)(blockIdx.y * 64 + srow) * K + sseg;

  const __bf16* brow[4];
#pragma unroll
  for (int nt = 0; nt < 4; ++nt)
    brow[nt] = Bt + (size_t)(n0 + nt * 16 + ln16) * K;

  auto stageA = [&](int buf, int kk2) {
#if USE_ASYNC_LDS
    __builtin_amdgcn_global_load_async_to_lds_b128(
        (__attribute__((address_space(1))) v4i*)(aStage + kk2),
        (__attribute__((address_space(3))) v4i*)&atile[buf][srow][sseg], 0, 0);
#else
    *reinterpret_cast<v8bf*>(&atile[buf][srow][sseg]) =
        *reinterpret_cast<const v8bf*>(aStage + kk2);
#endif
  };
  auto syncStage = [&]() {
#if USE_ASYNC_LDS
    asm volatile("s_wait_asynccnt 0" ::: "memory");
#endif
    __syncthreads();
  };
  auto loadB = [&](v16bf bfr[4], int kk2) {
#pragma unroll
    for (int nt = 0; nt < 4; ++nt)
      bfr[nt] = *reinterpret_cast<const v16bf*>(brow[nt] + kk2 + khalf * 16);
  };

  const v8f vzero = {};
  v8f acc[2][4];
#pragma unroll
  for (int mt = 0; mt < 2; ++mt)
#pragma unroll
    for (int nt = 0; nt < 4; ++nt) acc[mt][nt] = vzero;

  auto compute = [&](int buf, const v16bf bfr[4]) {
    const __bf16* l0 = &atile[buf][wr * 32 + ln16][0];
    const __bf16* l1 = &atile[buf][wr * 32 + 16 + ln16][0];
    const v16bf af0 = SHUF16(*reinterpret_cast<const v8bf*>(l0 + khalf * 16),
                             *reinterpret_cast<const v8bf*>(l0 + khalf * 16 + 8));
    const v16bf af1 = SHUF16(*reinterpret_cast<const v8bf*>(l1 + khalf * 16),
                             *reinterpret_cast<const v8bf*>(l1 + khalf * 16 + 8));
#pragma unroll
    for (int nt = 0; nt < 4; ++nt) {
      acc[0][nt] = wmma_bf16(af0, bfr[nt], acc[0][nt]);
      acc[1][nt] = wmma_bf16(af1, bfr[nt], acc[1][nt]);
    }
  };

  v16bf b0[4], b1[4];
  stageA(0, 0);
  loadB(b0, 0);
  syncStage();

  for (int kk = 0; kk < K; kk += 64) {  // K % 64 == 0
    // phase 0: prefetch kk+32, compute kk
    stageA(1, kk + 32);
    loadB(b1, kk + 32);
    compute(0, b0);
    syncStage();
    // phase 1: prefetch kk+64 (wrap-guard on last step), compute kk+32
    const int k2 = (kk + 64 < K) ? kk + 64 : 0;
    stageA(0, k2);
    loadB(b0, k2);
    compute(1, b1);
    syncStage();
  }

#pragma unroll
  for (int mt = 0; mt < 2; ++mt)
#pragma unroll
    for (int nt = 0; nt < 4; ++nt)
#pragma unroll
      for (int r = 0; r < 8; ++r) {
        const size_t off =
            (size_t)(m0 + mt * 16 + r + khalf * 8) * N + n0 + nt * 16 + ln16;
        if constexpr (OUT_BF16)
          reinterpret_cast<__bf16*>(Cout)[off] = f2bf(acc[mt][nt][r]);
        else
          reinterpret_cast<float*>(Cout)[off] = acc[mt][nt][r];
      }
}

// ---------------------------------------------------------------------------
// RoPE in place on bf16 buf laid out as (B*T, nheads, D); math in f32.
// ---------------------------------------------------------------------------
__global__ void rope_bf16(__bf16* __restrict__ buf, const float* __restrict__ cosb,
                          const float* __restrict__ sinb, int nheads, int total) {
  int idx = blockIdx.x * blockDim.x + threadIdx.x;
  if (idx >= total) return;
  int i    = idx & 63;
  int rem  = idx >> 6;
  int head = rem % nheads;
  int bt   = rem / nheads;
  int t    = bt & (T_ - 1);
  float c = cosb[t * 64 + i], s = sinb[t * 64 + i];
  __bf16* p = buf + ((size_t)bt * nheads + head) * D_;
  float x1 = bf2f(p[i]), x2 = bf2f(p[i + 64]);
  p[i]      = f2bf(x1 * c - x2 * s);
  p[i + 64] = f2bf(x2 * c + x1 * s);
}

// ---------------------------------------------------------------------------
// Per-head lambda scalar.
// ---------------------------------------------------------------------------
__global__ void lambda_kernel(const float* __restrict__ q1, const float* __restrict__ k1,
                              const float* __restrict__ q2, const float* __restrict__ k2,
                              float* __restrict__ lam) {
  int h = threadIdx.x;
  if (h >= H_) return;
  float s1 = 0.f, s2 = 0.f;
  for (int d = 0; d < D_; ++d) {
    s1 += q1[h * D_ + d] * k1[h * D_ + d];
    s2 += q2[h * D_ + d] * k2[h * D_ + d];
  }
  lam[h] = __expf(s1) - __expf(s2) + LAMBDA_INIT_;
}

// ---------------------------------------------------------------------------
// Fused differential flash attention (all-bf16 operands, f32 softmax state).
// Block = 128 threads = 4 waves, each wave one 16-row query tile. Per 32-key
// iteration: 16 score WMMAs + 16 PV WMMAs; every fragment is a contiguous
// b128 vector load; row reductions are permlane16 VALU butterflies; P is
// transposed C->A layout via per-wave LDS with explicit s_wait_dscnt (no
// block barriers: waves have different causal trip counts).
// ---------------------------------------------------------------------------
__global__ __launch_bounds__(128) void diff_attn_kernel(
    const __bf16* __restrict__ qf, const __bf16* __restrict__ kf,
    const __bf16* __restrict__ vT, const float* __restrict__ lamb,
    const float* __restrict__ sublnw, __bf16* __restrict__ out) {
  __shared__ __bf16 plds[4][16][32];

  const int tid   = threadIdx.x;
  const int lane  = tid & 31;
  const int ln16  = lane & 15;
  const int khalf = lane >> 4;
  const int wave  = tid >> 5;

  const int tileBlk = blockIdx.x & 15;
  const int bh = blockIdx.x >> 4;
  const int b = bh / H_;
  const int h = bh % H_;
  const int qbase = (tileBlk * 4 + wave) * 16;

  const int kh1 = h / NREP_;
  const int kh2 = HK_ + h / NREP_;
  const __bf16* vTbase = vT + (size_t)(b * HK_ + h / NREP_) * D_ * T_;

  v16bf aq1[4], aq2[4];
  {
    const __bf16* q1row = qf + ((size_t)(b * T_ + qbase + ln16) * (2 * H_) + h) * D_;
    const __bf16* q2row =
        qf + ((size_t)(b * T_ + qbase + ln16) * (2 * H_) + (H_ + h)) * D_;
#pragma unroll
    for (int c = 0; c < 4; ++c) {
      v8bf lo, hi;
      lo = *reinterpret_cast<const v8bf*>(q1row + 32 * c + khalf * 16);
      hi = *reinterpret_cast<const v8bf*>(q1row + 32 * c + khalf * 16 + 8);
      aq1[c] = SHUF16(lo, hi);
      lo = *reinterpret_cast<const v8bf*>(q2row + 32 * c + khalf * 16);
      hi = *reinterpret_cast<const v8bf*>(q2row + 32 * c + khalf * 16 + 8);
      aq2[c] = SHUF16(lo, hi);
    }
  }

  const v8f vzero = {};
  float m1[8], l1[8], m2[8], l2[8];
  v8f acc1[8], acc2[8];
#pragma unroll
  for (int r = 0; r < 8; ++r) { m1[r] = -1e30f; m2[r] = -1e30f; l1[r] = 0.f; l2[r] = 0.f; }
#pragma unroll
  for (int cc = 0; cc < 8; ++cc) { acc1[cc] = vzero; acc2[cc] = vzero; }

  const size_t krowstride = (size_t)(2 * HK_) * D_;  // elems per key step
  const int jend = (qbase + 15) >> 5;
  for (int j = 0; j <= jend; ++j) {
    const int kbase = j * 32;

    const __bf16* k1r0 = kf + ((size_t)(b * T_ + kbase + ln16) * (2 * HK_) + kh1) * D_;
    const __bf16* k1r1 =
        kf + ((size_t)(b * T_ + kbase + 16 + ln16) * (2 * HK_) + kh1) * D_;
    const __bf16* k2r0 = kf + ((size_t)(b * T_ + kbase + ln16) * (2 * HK_) + kh2) * D_;
    const __bf16* k2r1 =
        kf + ((size_t)(b * T_ + kbase + 16 + ln16) * (2 * HK_) + kh2) * D_;

    if (j < jend) {  // stream-ahead for the next key block (global_prefetch_b8)
      __builtin_prefetch(k1r0 + 32 * krowstride, 0, 1);
      __builtin_prefetch(k1r1 + 32 * krowstride, 0, 1);
      __builtin_prefetch(k2r0 + 32 * krowstride, 0, 1);
      __builtin_prefetch(k2r1 + 32 * krowstride, 0, 1);
    }

    v8f s10 = vzero, s11 = vzero, s20 = vzero, s21 = vzero;
#pragma unroll
    for (int c = 0; c < 4; ++c) {
      const int o = 32 * c + khalf * 16;
      s10 = wmma_bf16(aq1[c], *reinterpret_cast<const v16bf*>(k1r0 + o), s10);
      s11 = wmma_bf16(aq1[c], *reinterpret_cast<const v16bf*>(k1r1 + o), s11);
      s20 = wmma_bf16(aq2[c], *reinterpret_cast<const v16bf*>(k2r0 + o), s20);
      s21 = wmma_bf16(aq2[c], *reinterpret_cast<const v16bf*>(k2r1 + o), s21);
    }

    float so1[8], so2[8];
#pragma unroll
    for (int r = 0; r < 8; ++r) {
      const int qrow = qbase + r + khalf * 8;
      const int key0 = kbase + ln16;
      const int key1 = key0 + 16;
      float a0  = (key0 <= qrow) ? s10[r] * SCALE_ : -1e30f;
      float a1v = (key1 <= qrow) ? s11[r] * SCALE_ : -1e30f;
      float b0  = (key0 <= qrow) ? s20[r] * SCALE_ : -1e30f;
      float b1  = (key1 <= qrow) ? s21[r] * SCALE_ : -1e30f;
      const float mx1 = redmax16(fmaxf(a0, a1v));
      const float mx2 = redmax16(fmaxf(b0, b1));
      float mn1 = fmaxf(m1[r], mx1), mn2 = fmaxf(m2[r], mx2);
      so1[r] = __expf(m1[r] - mn1);
      so2[r] = __expf(m2[r] - mn2);
      m1[r] = mn1; m2[r] = mn2;
      float p0 = __expf(a0 - mn1), p1 = __expf(a1v - mn1);
      float q0 = __expf(b0 - mn2), q1v = __expf(b1 - mn2);
      const float rs1 = redsum16(p0 + p1);
      const float rs2 = redsum16(q0 + q1v);
      l1[r] = l1[r] * so1[r] + rs1;
      l2[r] = l2[r] * so2[r] + rs2;
      s10[r] = p0; s11[r] = p1; s20[r] = q0; s21[r] = q1v;
    }
#pragma unroll
    for (int cc = 0; cc < 8; ++cc)
#pragma unroll
      for (int r = 0; r < 8; ++r) {
        acc1[cc][r] *= so1[r];
        acc2[cc][r] *= so2[r];
      }

    v16bf p1A, p2A;
    {
#pragma unroll
      for (int r = 0; r < 8; ++r) {
        plds[wave][r + khalf * 8][ln16]      = f2bf(s10[r]);
        plds[wave][r + khalf * 8][ln16 + 16] = f2bf(s11[r]);
      }
      asm volatile("s_wait_dscnt 0" ::: "memory");
      const __bf16* prow = &plds[wave][ln16][0];
      v8bf lo = *reinterpret_cast<const v8bf*>(prow + khalf * 16);
      v8bf hi = *reinterpret_cast<const v8bf*>(prow + khalf * 16 + 8);
      p1A = SHUF16(lo, hi);
      asm volatile("s_wait_dscnt 0" ::: "memory");
#pragma unroll
      for (int r = 0; r < 8; ++r) {
        plds[wave][r + khalf * 8][ln16]      = f2bf(s20[r]);
        plds[wave][r + khalf * 8][ln16 + 16] = f2bf(s21[r]);
      }
      asm volatile("s_wait_dscnt 0" ::: "memory");
      lo = *reinterpret_cast<const v8bf*>(prow + khalf * 16);
      hi = *reinterpret_cast<const v8bf*>(prow + khalf * 16 + 8);
      p2A = SHUF16(lo, hi);
      asm volatile("s_wait_dscnt 0" ::: "memory");
    }

#pragma unroll
    for (int cc = 0; cc < 8; ++cc) {
      const __bf16* vptr =
          vTbase + (size_t)(cc * 16 + ln16) * T_ + kbase + khalf * 16;
      const v16bf bv = *reinterpret_cast<const v16bf*>(vptr);
      if (j < jend) __builtin_prefetch(vptr + 32, 0, 1);
      acc1[cc] = wmma_bf16(p1A, bv, acc1[cc]);
      acc2[cc] = wmma_bf16(p2A, bv, acc2[cc]);
    }
  }

  const float lam = lamb[h];
  float diff[8][8];
  float ssq[8];
#pragma unroll
  for (int r = 0; r < 8; ++r) ssq[r] = 0.f;
#pragma unroll
  for (int cc = 0; cc < 8; ++cc)
#pragma unroll
    for (int r = 0; r < 8; ++r) {
      float a1 = acc1[cc][r] / l1[r];
      float a2 = acc2[cc][r] / l2[r];
      float dv = a1 - lam * a2;
      diff[cc][r] = dv;
      ssq[r] += dv * dv;
    }
#pragma unroll
  for (int r = 0; r < 8; ++r) {
    ssq[r] = redsum16(ssq[r]);
    ssq[r] = rsqrtf(ssq[r] * (1.0f / D_) + EPS_);
  }
#pragma unroll
  for (int cc = 0; cc < 8; ++cc) {
    const float w = sublnw[cc * 16 + ln16];
#pragma unroll
    for (int r = 0; r < 8; ++r) {
      const int trow = qbase + r + khalf * 8;
      out[(size_t)(b * T_ + trow) * (H_ * D_) + h * D_ + cc * 16 + ln16] =
          f2bf(diff[cc][r] * ssq[r] * w);
    }
  }
}

// ---------------------------------------------------------------------------
extern "C" void kernel_launch(void* const* d_in, const int* in_sizes, int n_in,
                              void* d_out, int out_size, void* d_ws, size_t ws_size,
                              hipStream_t stream) {
  const float* x    = (const float*)d_in[0];
  const float* cosb = (const float*)d_in[1];
  const float* sinb = (const float*)d_in[2];
  const float* wq   = (const float*)d_in[3];
  const float* wk   = (const float*)d_in[4];
  const float* wv   = (const float*)d_in[5];
  const float* wo   = (const float*)d_in[6];
  const float* lq1  = (const float*)d_in[7];
  const float* lk1  = (const float*)d_in[8];
  const float* lq2  = (const float*)d_in[9];
  const float* lk2  = (const float*)d_in[10];
  const float* slw  = (const float*)d_in[11];
  float* out = (float*)d_out;

  const size_t M = (size_t)B_ * T_;  // 2048
  char* ws = (char*)d_ws;
  __bf16* x_bf   = (__bf16*)ws;                 ws += M * DIM_ * 2;            // 16 MB
  __bf16* wqT    = (__bf16*)ws;                 ws += (size_t)8192 * 4096 * 2; // 64 MB
  __bf16* wkT    = (__bf16*)ws;                 ws += (size_t)2048 * 4096 * 2; // 16 MB
  __bf16* wvT    = (__bf16*)ws;                 ws += (size_t)1024 * 4096 * 2; //  8 MB
  __bf16* woT    = (__bf16*)ws;                 ws += (size_t)4096 * 4096 * 2; // 32 MB
  __bf16* q_bf   = (__bf16*)ws;                 ws += M * 2 * H_ * D_ * 2;     // 32 MB
  __bf16* k_bf   = (__bf16*)ws;                 ws += M * 2 * HK_ * D_ * 2;    //  8 MB
  __bf16* v_bf   = (__bf16*)ws;                 ws += M * HK_ * D_ * 2;        //  4 MB
  __bf16* vT_bf  = (__bf16*)ws;                 ws += M * HK_ * D_ * 2;        //  4 MB
  __bf16* att_bf = (__bf16*)ws;                 ws += M * H_ * D_ * 2;         // 16 MB
  float*  lam    = (float*)ws;

  {
    int t4 = (int)(M * DIM_ / 4);
    convert_bf16<<<(t4 + 255) / 256, 256, 0, stream>>>(x, x_bf, t4);
  }
  transpose_convert<<<dim3(8192 / 32, DIM_ / 32), dim3(32, 8), 0, stream>>>(wq, wqT, DIM_, 8192);
  transpose_convert<<<dim3(2048 / 32, DIM_ / 32), dim3(32, 8), 0, stream>>>(wk, wkT, DIM_, 2048);
  transpose_convert<<<dim3(1024 / 32, DIM_ / 32), dim3(32, 8), 0, stream>>>(wv, wvT, DIM_, 1024);
  transpose_convert<<<dim3(4096 / 32, 4096 / 32), dim3(32, 8), 0, stream>>>(wo, woT, 4096, DIM_);

  gemm_bf16_wmma<true><<<dim3(8192 / 256, M / 64), 256, 0, stream>>>(
      x_bf, wqT, q_bf, (int)M, 8192, DIM_);
  gemm_bf16_wmma<true><<<dim3(2048 / 256, M / 64), 256, 0, stream>>>(
      x_bf, wkT, k_bf, (int)M, 2048, DIM_);
  gemm_bf16_wmma<true><<<dim3(1024 / 256, M / 64), 256, 0, stream>>>(
      x_bf, wvT, v_bf, (int)M, 1024, DIM_);

  {
    int totq = (int)(M * 2 * H_ * (D_ / 2));
    rope_bf16<<<(totq + 255) / 256, 256, 0, stream>>>(q_bf, cosb, sinb, 2 * H_, totq);
    int totk = (int)(M * 2 * HK_ * (D_ / 2));
    rope_bf16<<<(totk + 255) / 256, 256, 0, stream>>>(k_bf, cosb, sinb, 2 * HK_, totk);
  }
  transpose_v<<<dim3(D_ / 32, T_ / 32, B_ * HK_), dim3(32, 8), 0, stream>>>(v_bf, vT_bf);
  lambda_kernel<<<1, 32, 0, stream>>>(lq1, lk1, lq2, lk2, lam);

  diff_attn_kernel<<<B_ * H_ * (T_ / 64), 128, 0, stream>>>(
      q_bf, k_bf, vT_bf, lam, slw, att_bf);

  gemm_bf16_wmma<false><<<dim3(DIM_ / 256, M / 64), 256, 0, stream>>>(
      att_bf, woT, out, (int)M, DIM_, H_ * D_);
}
